// extractor_78735340470835
// MI455X (gfx1250) — matrix operand order repaired and checked by reading, not compiled, not run on hardware
//
#include <hip/hip_runtime.h>
#include <math.h>

// ---------------------------------------------------------------------------
// Problem constants (from the reference)
// ---------------------------------------------------------------------------
constexpr int D   = 384;
constexpr int H   = 6;
constexpr int DK  = 64;
constexpr int HID = 96;
constexpr int NS  = 2040;
constexpr int NP  = 2048;
constexpr int BB  = 8;
constexpr int MQ  = BB * NS;   // 16320 query rows
constexpr int MK  = BB * NP;   // 16384 key rows

typedef __attribute__((ext_vector_type(16))) _Float16 v16h;
typedef __attribute__((ext_vector_type(8)))  float    v8f;

#if defined(__HIP_DEVICE_COMPILE__) && \
    __has_builtin(__builtin_amdgcn_global_load_async_to_lds_b128) && \
    __has_builtin(__builtin_amdgcn_s_wait_asynccnt)
#define USE_ASYNC_LDS 1
typedef int v4i_async __attribute__((vector_size(16)));
typedef v4i_async __attribute__((address_space(1)))* gas_v4i;   // global (AS1)
typedef v4i_async __attribute__((address_space(3)))* las_v4i;   // LDS (AS3)
#else
#define USE_ASYNC_LDS 0
#endif

union FragH {
    v16h     v;
    uint4    q[2];
    _Float16 e[16];
};

__device__ inline v8f wmma_f16(v16h a, v16h b, v8f c) {
    // D = A(16x32) * B(32x16) + C  -> v_wmma_f32_16x16x32_f16
    return __builtin_amdgcn_wmma_f32_16x16x32_f16(
        /*neg_a=*/false, a, /*neg_b=*/false, b,
        /*c_mod=*/(short)0, c, /*reuse_a=*/false, /*reuse_b=*/false);
}

__device__ inline v8f v8f_zero() {
    v8f z = {0.f, 0.f, 0.f, 0.f, 0.f, 0.f, 0.f, 0.f};
    return z;
}

// Paired full-wave (32 lane) sum: reduces two values in one interleaved chain
// so each shuffle stage shares a single dscnt wait.
__device__ inline void wave_sum2(float& a, float& b) {
#pragma unroll
    for (int o = 16; o > 0; o >>= 1) {
        float ta = __shfl_xor(a, o, 32);
        float tb = __shfl_xor(b, o, 32);
        a += ta;
        b += tb;
    }
}

// A fragment (16x32 f16, row-major source, leading dim ld):
// lane l (l<16): row l, K = 0..15 contiguous; lanes 16..31: K = 16..31
__device__ inline v16h load_frag_a(const _Float16* __restrict__ tile, int ld, int lane) {
    const int m  = lane & 15;
    const int kb = (lane < 16) ? 0 : 16;
    const _Float16* p = tile + (size_t)m * ld + kb;
    FragH f;
    f.q[0] = *reinterpret_cast<const uint4*>(p);
    f.q[1] = *reinterpret_cast<const uint4*>(p + 8);
    return f.v;
}

// B fragment (32x16 f16) from an N x K row-major weight matrix, tile at [n0][k0]:
// lane l (l<16): column l, K = 0..15 contiguous; lanes 16..31: K = 16..31
__device__ inline v16h load_frag_b(const _Float16* __restrict__ tile, int ld, int lane) {
    const int n  = lane & 15;
    const int ko = (lane < 16) ? 0 : 16;
    const _Float16* p = tile + (size_t)n * ld + ko;
    FragH f;
    f.q[0] = *reinterpret_cast<const uint4*>(p);
    f.q[1] = *reinterpret_cast<const uint4*>(p + 8);
    return f.v;
}

// ---------------------------------------------------------------------------
// fp32 -> fp16 weight conversion
// ---------------------------------------------------------------------------
__global__ void cvt_f32_to_f16(const float* __restrict__ in, _Float16* __restrict__ out, int n) {
    int i = blockIdx.x * blockDim.x + threadIdx.x;
    if (i < n) out[i] = (_Float16)in[i];
}

// ---------------------------------------------------------------------------
// Double LayerNorm (eps1 then eps2), fp32 rows of 384 -> fp16 rows. Wave/row.
// Uses E[x^2]-m^2 so mean+var reduce in ONE paired shuffle chain per stage.
// ---------------------------------------------------------------------------
__global__ __launch_bounds__(256) void ln2_to_h(
    const float* __restrict__ in,
    const float* __restrict__ g1, const float* __restrict__ b1,
    const float* __restrict__ g2, const float* __restrict__ b2,
    _Float16* __restrict__ out, int rows, float eps1, float eps2) {
    const int row = blockIdx.x * 8 + (threadIdx.x >> 5);
    if (row >= rows) return;
    const int lane = threadIdx.x & 31;
    const float* p = in + (size_t)row * D;

    float x[12];
#pragma unroll
    for (int j = 0; j < 12; ++j) x[j] = p[lane + 32 * j];

    float s1 = 0.f, s2 = 0.f;
#pragma unroll
    for (int j = 0; j < 12; ++j) { s1 += x[j]; s2 += x[j] * x[j]; }
    wave_sum2(s1, s2);
    const float m1 = s1 * (1.0f / 384.0f);
    const float r1 = rsqrtf(fmaxf(s2 * (1.0f / 384.0f) - m1 * m1, 0.f) + eps1);

    float y[12];
#pragma unroll
    for (int j = 0; j < 12; ++j) {
        int c = lane + 32 * j;
        y[j] = (x[j] - m1) * r1 * g1[c] + b1[c];
    }
    s1 = 0.f; s2 = 0.f;
#pragma unroll
    for (int j = 0; j < 12; ++j) { s1 += y[j]; s2 += y[j] * y[j]; }
    wave_sum2(s1, s2);
    const float m2 = s1 * (1.0f / 384.0f);
    const float r2 = rsqrtf(fmaxf(s2 * (1.0f / 384.0f) - m2 * m2, 0.f) + eps2);

    _Float16* o = out + (size_t)row * D;
#pragma unroll
    for (int j = 0; j < 12; ++j) {
        int c = lane + 32 * j;
        o[c] = (_Float16)((y[j] - m2) * r2 * g2[c] + b2[c]);
    }
}

// ---------------------------------------------------------------------------
// Generic WMMA GEMM: C(MxN) = A(MxK,f16,rowmajor) @ W(NxK,f16,rowmajor)^T + bias
// Block = 128 threads = 4 waves; wave -> one 16x64 output tile (all 4 waves
// share one 64x32 weight tile, staged in LDS via async-to-LDS DMA when the
// toolchain exposes the gfx1250 builtins).
// Requires M % 64 == 0 (all call sites satisfy this).
// Store MODEs: 0=f32, 1=f16, 2=f32 + f16*0.125 (Q: fold score scale into f16 copy),
//              3=f16 transposed (V^T, per-batch NP=2048),
//              4=exact GELU then f16, 5=f32 with residual add.
// ---------------------------------------------------------------------------
template <int MODE>
__global__ __launch_bounds__(128) void gemm16(
    const _Float16* __restrict__ A, const _Float16* __restrict__ W,
    const float* __restrict__ bias, float* __restrict__ Cf,
    _Float16* __restrict__ Ch, const float* __restrict__ resid,
    int M, int N, int K) {
    const int lane = threadIdx.x & 31;
    const int wave = threadIdx.x >> 5;
    const int m0 = (blockIdx.x * 4 + wave) * 16;
    const int n0g = blockIdx.y * 64;

#if USE_ASYNC_LDS
    constexpr int LDSW = 40;                 // padded row stride (halfs)
    __shared__ _Float16 wtile[64 * LDSW];    // 5 KB weight tile
#endif

    v8f acc[4];
#pragma unroll
    for (int s = 0; s < 4; ++s) acc[s] = v8f_zero();

    for (int k0 = 0; k0 < K; k0 += 32) {
#if USE_ASYNC_LDS
        // Async DMA the 64x32 weight tile into LDS (2 x 16B chunks / thread).
#pragma unroll
        for (int c = 0; c < 2; ++c) {
            const int chunk = threadIdx.x + c * 128;   // 0..255
            const int n  = chunk >> 2;                 // tile row 0..63
            const int cc = chunk & 3;                  // 16B chunk in row
            int nsrc = n0g + n;
            if (nsrc >= N) nsrc = N - 1;               // clamp (never consumed)
            const _Float16* g = W + (size_t)nsrc * K + k0 + cc * 8;
            _Float16* l = wtile + n * LDSW + cc * 8;
            __builtin_amdgcn_global_load_async_to_lds_b128(
                (gas_v4i)(uintptr_t)g,   // inttoptr: global VA == AS1 VA
                (las_v4i)l,              // addrspacecast generic -> LDS
                0, 0);
        }
        __builtin_amdgcn_s_wait_asynccnt(0);
        __syncthreads();
#endif
        if (k0 + 32 < K)
            __builtin_prefetch(A + (size_t)m0 * K + k0 + 32, 0, 3);
        v16h a = load_frag_a(A + (size_t)m0 * K + k0, K, lane);
#pragma unroll
        for (int s = 0; s < 4; ++s) {
            const int n0 = n0g + s * 16;
            if (n0 < N) {
#if USE_ASYNC_LDS
                const int ko = (lane < 16) ? 0 : 16;
                const _Float16* p = wtile + (s * 16 + (lane & 15)) * LDSW + ko;
                FragH f;
                f.q[0] = *reinterpret_cast<const uint4*>(p);
                f.q[1] = *reinterpret_cast<const uint4*>(p + 8);
                acc[s] = wmma_f16(a, f.v, acc[s]);
#else
                v16h b = load_frag_b(W + (size_t)n0 * K + k0, K, lane);
                acc[s] = wmma_f16(a, b, acc[s]);
#endif
            }
        }
#if USE_ASYNC_LDS
        __syncthreads();   // protect wtile before next step's DMA
#endif
    }

#pragma unroll
    for (int s = 0; s < 4; ++s) {
        const int n0 = n0g + s * 16;
        if (n0 >= N) continue;
        const int col = n0 + (lane & 15);
        const float bcol = bias[col];
#pragma unroll
        for (int i = 0; i < 8; ++i) {
            const int row = m0 + ((lane < 16) ? i : 8 + i);
            const float val = acc[s][i] + bcol;
            const size_t idx = (size_t)row * N + col;
            if (MODE == 0) Cf[idx] = val;
            if (MODE == 1) Ch[idx] = (_Float16)val;
            if (MODE == 2) { Cf[idx] = val; Ch[idx] = (_Float16)(val * 0.125f); }
            if (MODE == 3) {
                // V^T layout: [b][col(384)][key(2048)]  (row = b*2048 + key)
                Ch[((size_t)(row >> 11) * D + col) * (size_t)NP + (row & 2047)] = (_Float16)val;
            }
            if (MODE == 4) {
                const float g = 0.5f * val * (1.0f + erff(val * 0.70710678118654752f));
                Ch[idx] = (_Float16)g;
            }
            if (MODE == 5) Cf[idx] = val + resid[idx];
        }
    }
}

// ---------------------------------------------------------------------------
// Attention pass 1: per (batch, 16-query tile), 6 waves (1/head).
// Streams keys in 32-wide steps with WMMA Q.K^T (Q pre-scaled by 1/8).
// PER-LANE online (max, sumexp) stats -> zero cross-lane traffic in the loop;
// one 4-stage flash-merge across the 16-lane half at the end.
// ---------------------------------------------------------------------------
__global__ __launch_bounds__(192) void attn_pass1(
    const _Float16* __restrict__ qh, const _Float16* __restrict__ kh,
    float* __restrict__ sm, float* __restrict__ sz) {
    const int b = blockIdx.y;
    const int qt = blockIdx.x;
    const int h = threadIdx.x >> 5;
    const int lane = threadIdx.x & 31;
    const int mrow = lane & 15;
    const int kb  = (lane < 16) ? 0 : 16;
    const int dko = (lane < 16) ? 0 : 16;

    int gq = qt * 16 + mrow;
    if (gq >= NS) gq = NS - 1;  // clamp (duplicate) rows in the ragged last tile

    FragH qa[2];
    const _Float16* qbase = qh + (size_t)(b * NS + gq) * D + h * DK;
#pragma unroll
    for (int ch = 0; ch < 2; ++ch) {
        const _Float16* p = qbase + ch * 32 + kb;
        qa[ch].q[0] = *reinterpret_cast<const uint4*>(p);
        qa[ch].q[1] = *reinterpret_cast<const uint4*>(p + 8);
    }

    float mrun[8], lrun[8];
#pragma unroll
    for (int i = 0; i < 8; ++i) { mrun[i] = -1e30f; lrun[i] = 0.f; }

    for (int kt = 0; kt < NP; kt += 32) {
        if (kt + 32 < NP)   // prefetch next key tile (global_prefetch_b8)
            __builtin_prefetch(kh + (size_t)(b * NP + kt + 32 + mrow) * D + h * DK, 0, 3);
        v8f s0 = v8f_zero(), s1 = v8f_zero();
#pragma unroll
        for (int sub = 0; sub < 2; ++sub) {
            const int key = kt + sub * 16 + mrow;  // B-frag column
            const _Float16* kbp = kh + (size_t)(b * NP + key) * D + h * DK + dko;
#pragma unroll
            for (int ch = 0; ch < 2; ++ch) {
                FragH kf;
                const _Float16* p = kbp + ch * 32;
                kf.q[0] = *reinterpret_cast<const uint4*>(p);
                kf.q[1] = *reinterpret_cast<const uint4*>(p + 8);
                if (sub == 0) s0 = wmma_f16(qa[ch].v, kf.v, s0);
                else          s1 = wmma_f16(qa[ch].v, kf.v, s1);
            }
        }
        // Pure-VALU per-lane online update (co-executes with the WMMA pipe)
#pragma unroll
        for (int i = 0; i < 8; ++i) {
            const float a0 = s0[i];
            const float a1 = s1[i];
            const float mnew = fmaxf(mrun[i], fmaxf(a0, a1));
            lrun[i] = lrun[i] * __expf(mrun[i] - mnew)
                    + __expf(a0 - mnew) + __expf(a1 - mnew);
            mrun[i] = mnew;
        }
    }

    // Flash-style (m,l) merge across the 16 lanes of each half (once).
#pragma unroll
    for (int o = 8; o > 0; o >>= 1) {
#pragma unroll
        for (int i = 0; i < 8; ++i) {
            const float om = __shfl_xor(mrun[i], o, 32);
            const float ol = __shfl_xor(lrun[i], o, 32);
            const float mn = fmaxf(mrun[i], om);
            lrun[i] = lrun[i] * __expf(mrun[i] - mn) + ol * __expf(om - mn);
            mrun[i] = mn;
        }
    }

    if ((lane & 15) == 0) {
#pragma unroll
        for (int i = 0; i < 8; ++i) {
            const int r = (lane < 16) ? i : 8 + i;
            const int q = qt * 16 + r;
            if (q < NS) {
                const size_t idx = (size_t)(b * H + h) * NS + q;
                sm[idx] = mrun[i];
                sz[idx] = 1.0f / lrun[i];
            }
        }
    }
}

// ---------------------------------------------------------------------------
// Attention pass 2: recompute probs, cross-head renormalize via LDS,
// accumulate O = P.V with WMMA, emit d = q - o (fp16) for the Wc GEMM.
// ---------------------------------------------------------------------------
__global__ __launch_bounds__(192) void attn_pass2(
    const _Float16* __restrict__ qh, const _Float16* __restrict__ kh,
    const _Float16* __restrict__ vt, const float* __restrict__ qf,
    const float* __restrict__ sm, const float* __restrict__ sz,
    _Float16* __restrict__ dh) {
    __shared__ float lds_p[H][16][32];
    __shared__ float lds_rden[16][32];

    const int b = blockIdx.y;
    const int qt = blockIdx.x;
    const int h = threadIdx.x >> 5;
    const int lane = threadIdx.x & 31;
    const int mrow = lane & 15;
    const int kb  = (lane < 16) ? 0 : 16;
    const int dko = (lane < 16) ? 0 : 16;

    int gq = qt * 16 + mrow;
    if (gq >= NS) gq = NS - 1;

    FragH qa[2];
    const _Float16* qbase = qh + (size_t)(b * NS + gq) * D + h * DK;
#pragma unroll
    for (int ch = 0; ch < 2; ++ch) {
        const _Float16* p = qbase + ch * 32 + kb;
        qa[ch].q[0] = *reinterpret_cast<const uint4*>(p);
        qa[ch].q[1] = *reinterpret_cast<const uint4*>(p + 8);
    }

    // Per-C-layout-row softmax stats
    float m8[8], z8[8];
#pragma unroll
    for (int i = 0; i < 8; ++i) {
        const int r = (lane < 16) ? i : 8 + i;
        int q = qt * 16 + r;
        if (q >= NS) q = NS - 1;
        const size_t idx = (size_t)(b * H + h) * NS + q;
        m8[i] = sm[idx];
        z8[i] = sz[idx];
    }

    v8f oacc[4];
#pragma unroll
    for (int s = 0; s < 4; ++s) oacc[s] = v8f_zero();

    for (int kt = 0; kt < NP; kt += 32) {
        if (kt + 32 < NP) {  // prefetch next K and V tiles
            __builtin_prefetch(kh + (size_t)(b * NP + kt + 32 + mrow) * D + h * DK, 0, 3);
            __builtin_prefetch(vt + ((size_t)b * D + h * DK + mrow) * (size_t)NP + kt + 32, 0, 3);
        }
        v8f s0 = v8f_zero(), s1 = v8f_zero();
#pragma unroll
        for (int sub = 0; sub < 2; ++sub) {
            const int key = kt + sub * 16 + mrow;
            const _Float16* kbp = kh + (size_t)(b * NP + key) * D + h * DK + dko;
#pragma unroll
            for (int ch = 0; ch < 2; ++ch) {
                FragH kf;
                const _Float16* p = kbp + ch * 32;
                kf.q[0] = *reinterpret_cast<const uint4*>(p);
                kf.q[1] = *reinterpret_cast<const uint4*>(p + 8);
                if (sub == 0) s0 = wmma_f16(qa[ch].v, kf.v, s0);
                else          s1 = wmma_f16(qa[ch].v, kf.v, s1);
            }
        }
        // True per-head attention probs -> LDS (Q already carries the 1/8 scale)
#pragma unroll
        for (int i = 0; i < 8; ++i) {
            const int r = (lane < 16) ? i : 8 + i;
            lds_p[h][r][mrow]      = __expf(s0[i] - m8[i]) * z8[i];
            lds_p[h][r][mrow + 16] = __expf(s1[i] - m8[i]) * z8[i];
        }
        __syncthreads();
        // Reciprocal cross-head denominator (512 elements / 192 threads)
        for (int e = threadIdx.x; e < 512; e += 192) {
            const int r = e >> 5, c = e & 31;
            float dsum = 1e-9f;
#pragma unroll
            for (int hh = 0; hh < H; ++hh) dsum += lds_p[hh][r][c];
            lds_rden[r][c] = 1.0f / dsum;
        }
        __syncthreads();
        // Renormalized weights -> A fragment (f16), multiply-only
        FragH wf;
#pragma unroll
        for (int j = 0; j < 8; ++j) {
            wf.e[j]     = (_Float16)(lds_p[h][mrow][kb + j]      * lds_rden[mrow][kb + j]);
            wf.e[8 + j] = (_Float16)(lds_p[h][mrow][kb + 8 + j] * lds_rden[mrow][kb + 8 + j]);
        }
        // O += P @ V   (V stored transposed: [b][d(384)][key(2048)])
#pragma unroll
        for (int s = 0; s < 4; ++s) {
            const _Float16* p =
                vt + ((size_t)b * D + h * DK + s * 16 + mrow) * (size_t)NP + kt + dko;
            FragH vf;
            vf.q[0] = *reinterpret_cast<const uint4*>(p);
            vf.q[1] = *reinterpret_cast<const uint4*>(p + 8);
            oacc[s] = wmma_f16(wf.v, vf.v, oacc[s]);
        }
        __syncthreads();  // protect lds_p reuse next step
    }

    // d = q - o  (fp16, feeds the Wc GEMM)
#pragma unroll
    for (int s = 0; s < 4; ++s) {
        const int col = h * DK + s * 16 + mrow;
#pragma unroll
        for (int i = 0; i < 8; ++i) {
            const int r = (lane < 16) ? i : 8 + i;
            const int q = qt * 16 + r;
            if (q < NS) {
                const size_t idx = (size_t)(b * NS + q) * D + col;
                dh[idx] = (_Float16)(qf[idx] - oacc[s][i]);
            }
        }
    }
}

// ---------------------------------------------------------------------------
// Post-attention row kernel: r = LeakyReLU(LN(r_pre)); spm2 = spm + qf + r;
// hin = f16(LN(spm2, ffn)). Wave/row.
// ---------------------------------------------------------------------------
__global__ __launch_bounds__(256) void cross_row(
    const float* __restrict__ rpre, const float* __restrict__ qf,
    const float* __restrict__ spm,
    const float* __restrict__ gc, const float* __restrict__ bc,
    const float* __restrict__ gf, const float* __restrict__ bf,
    float* __restrict__ spm2, _Float16* __restrict__ hin, int rows) {
    const int row = blockIdx.x * 8 + (threadIdx.x >> 5);
    if (row >= rows) return;
    const int lane = threadIdx.x & 31;
    const size_t base = (size_t)row * D;

    float x[12];
#pragma unroll
    for (int j = 0; j < 12; ++j) x[j] = rpre[base + lane + 32 * j];
    float s1 = 0.f, s2 = 0.f;
#pragma unroll
    for (int j = 0; j < 12; ++j) { s1 += x[j]; s2 += x[j] * x[j]; }
    wave_sum2(s1, s2);
    const float m1 = s1 * (1.0f / 384.0f);
    const float r1 = rsqrtf(fmaxf(s2 * (1.0f / 384.0f) - m1 * m1, 0.f) + 1e-5f);

    float c[12];
#pragma unroll
    for (int j = 0; j < 12; ++j) {
        const int col = lane + 32 * j;
        float r = (x[j] - m1) * r1 * gc[col] + bc[col];
        r = (r > 0.f) ? r : 0.02f * r;                 // LeakyReLU(0.02)
        c[j] = spm[base + col] + qf[base + col] + r;   // spm + cross
        spm2[base + col] = c[j];
    }
    s1 = 0.f; s2 = 0.f;
#pragma unroll
    for (int j = 0; j < 12; ++j) { s1 += c[j]; s2 += c[j] * c[j]; }
    wave_sum2(s1, s2);
    const float m2 = s1 * (1.0f / 384.0f);
    const float r2 = rsqrtf(fmaxf(s2 * (1.0f / 384.0f) - m2 * m2, 0.f) + 1e-5f);
#pragma unroll
    for (int j = 0; j < 12; ++j) {
        const int col = lane + 32 * j;
        hin[base + col] = (_Float16)((c[j] - m2) * r2 * gf[col] + bf[col]);
    }
}

// ---------------------------------------------------------------------------
// Host orchestration
// ---------------------------------------------------------------------------
extern "C" void kernel_launch(void* const* d_in, const int* in_sizes, int n_in,
                              void* d_out, int out_size, void* d_ws, size_t ws_size,
                              hipStream_t stream) {
    const float* spm_feature = (const float*)d_in[0];
    const float* pct_feature = (const float*)d_in[1];
    const float* pct_g = (const float*)d_in[2];
    const float* pct_b = (const float*)d_in[3];
    const float* spm_g = (const float*)d_in[4];
    const float* spm_b = (const float*)d_in[5];
    const float* lnx_g = (const float*)d_in[6];
    const float* lnx_b = (const float*)d_in[7];
    const float* lny_g = (const float*)d_in[8];
    const float* lny_b = (const float*)d_in[9];
    const float* Wq = (const float*)d_in[10]; const float* bq  = (const float*)d_in[11];
    const float* Wk = (const float*)d_in[12]; const float* bk  = (const float*)d_in[13];
    const float* Wv = (const float*)d_in[14]; const float* bv  = (const float*)d_in[15];
    const float* Wc = (const float*)d_in[16]; const float* bc  = (const float*)d_in[17];
    const float* lnc_g = (const float*)d_in[18]; const float* lnc_b = (const float*)d_in[19];
    const float* ffn_g = (const float*)d_in[20]; const float* ffn_b = (const float*)d_in[21];
    const float* W1  = (const float*)d_in[22]; const float* fb1 = (const float*)d_in[23];
    const float* Wdw = (const float*)d_in[24]; const float* fbd = (const float*)d_in[25];
    const float* W2  = (const float*)d_in[26]; const float* fb2 = (const float*)d_in[27];

    char* ws = (char*)d_ws;
    size_t off = 0;
    auto alloc = [&](size_t bytes) -> char* {
        char* p = ws + off;
        off = (off + bytes + 255) & ~(size_t)255;
        return p;
    };

    _Float16* xn   = (_Float16*)alloc((size_t)MQ * D * 2);
    _Float16* yn   = (_Float16*)alloc((size_t)MK * D * 2);
    _Float16* q_h  = (_Float16*)alloc((size_t)MQ * D * 2);
    float*    q_f  = (float*)   alloc((size_t)MQ * D * 4);
    _Float16* k_h  = (_Float16*)alloc((size_t)MK * D * 2);
    _Float16* vT   = (_Float16*)alloc((size_t)MK * D * 2);
    float*    smax = (float*)   alloc((size_t)BB * H * NS * 4);
    float*    sinv = (float*)   alloc((size_t)BB * H * NS * 4);
    _Float16* dq   = (_Float16*)alloc((size_t)MQ * D * 2);
    float*    rpre = (float*)   alloc((size_t)MQ * D * 4);
    float*    spm2 = (float*)   alloc((size_t)MQ * D * 4);
    _Float16* hin  = (_Float16*)alloc((size_t)MQ * D * 2);
    _Float16* h1   = (_Float16*)alloc((size_t)MQ * HID * 2);
    _Float16* h2   = (_Float16*)alloc((size_t)MQ * HID * 2);
    _Float16* wq_h = (_Float16*)alloc((size_t)D * D * 2);
    _Float16* wk_h = (_Float16*)alloc((size_t)D * D * 2);
    _Float16* wv_h = (_Float16*)alloc((size_t)D * D * 2);
    _Float16* wc_h = (_Float16*)alloc((size_t)D * D * 2);
    _Float16* w1_h = (_Float16*)alloc((size_t)HID * D * 2);
    _Float16* wd_h = (_Float16*)alloc((size_t)HID * HID * 2);
    _Float16* w2_h = (_Float16*)alloc((size_t)D * HID * 2);
    (void)ws_size; (void)n_in; (void)in_sizes; (void)out_size;

    auto cvt = [&](const float* src, _Float16* dst, int n) {
        cvt_f32_to_f16<<<dim3((n + 255) / 256), dim3(256), 0, stream>>>(src, dst, n);
    };
    cvt(Wq, wq_h, D * D);
    cvt(Wk, wk_h, D * D);
    cvt(Wv, wv_h, D * D);
    cvt(Wc, wc_h, D * D);
    cvt(W1, w1_h, HID * D);
    cvt(Wdw, wd_h, HID * HID);
    cvt(W2, w2_h, D * HID);

    // Double LayerNorms -> fp16 activations
    ln2_to_h<<<dim3(MQ / 8), dim3(256), 0, stream>>>(
        spm_feature, spm_g, spm_b, lnx_g, lnx_b, xn, MQ, 1e-6f, 1e-5f);
    ln2_to_h<<<dim3(MK / 8), dim3(256), 0, stream>>>(
        pct_feature, pct_g, pct_b, lny_g, lny_b, yn, MK, 1e-6f, 1e-5f);

    // Projections: Q (f32 + pre-scaled f16), K (f16), V (f16 transposed)
    gemm16<2><<<dim3(MQ / 64, D / 64), dim3(128), 0, stream>>>(
        xn, wq_h, bq, q_f, q_h, nullptr, MQ, D, D);
    gemm16<1><<<dim3(MK / 64, D / 64), dim3(128), 0, stream>>>(
        yn, wk_h, bk, nullptr, k_h, nullptr, MK, D, D);
    gemm16<3><<<dim3(MK / 64, D / 64), dim3(128), 0, stream>>>(
        yn, wv_h, bv, nullptr, vT, nullptr, MK, D, D);

    // Attention (two-pass, cross-head renormalized)
    attn_pass1<<<dim3((NS + 15) / 16, BB), dim3(192), 0, stream>>>(q_h, k_h, smax, sinv);
    attn_pass2<<<dim3((NS + 15) / 16, BB), dim3(192), 0, stream>>>(
        q_h, k_h, vT, q_f, smax, sinv, dq);

    // r_pre = (q - o) @ Wc^T + bc
    gemm16<0><<<dim3(MQ / 64, D / 64), dim3(128), 0, stream>>>(
        dq, wc_h, bc, rpre, nullptr, nullptr, MQ, D, D);

    // LN + LeakyReLU + residuals + FFN pre-norm
    cross_row<<<dim3(MQ / 8), dim3(256), 0, stream>>>(
        rpre, q_f, spm_feature, lnc_g, lnc_b, ffn_g, ffn_b, spm2, hin, MQ);

    // FFN: fc1 -> shared 1x1 conv -> exact GELU -> fc2 (+ residual into d_out)
    gemm16<1><<<dim3(MQ / 64, 2), dim3(128), 0, stream>>>(
        hin, w1_h, fb1, nullptr, h1, nullptr, MQ, HID, D);
    gemm16<4><<<dim3(MQ / 64, 2), dim3(128), 0, stream>>>(
        h1, wd_h, fbd, nullptr, h2, nullptr, MQ, HID, HID);
    gemm16<5><<<dim3(MQ / 64, D / 64), dim3(128), 0, stream>>>(
        h2, w2_h, fb2, (float*)d_out, nullptr, spm2, MQ, D, HID);
}
